// GGNN_1108101562482
// MI455X (gfx1250) — matrix operand order repaired and checked by reading, not compiled, not run on hardware
//
#include <hip/hip_runtime.h>
#include <hip/hip_bf16.h>

// ---------------------------------------------------------------------------
// GGNN forward on MI455X (gfx1250, wave32, WMMA).
//
// Restructure: incoming[dst] = (sum_{edges->dst} u[src]) @ W2^T + deg(dst)*b2
// with u = leaky_relu(h @ W1^T + b1) computed per NODE (not per edge).
// All GEMMs use v_wmma_f32_16x16x32_f16 (f16 A/B, f32 accum), K padded to 160.
// Each wave owns a 32-row strip (2 A-register sets) -> 10 WMMAs per B tile;
// B loads are bulk-issued (sched_barrier) so each tile pays one load-wait.
// Scatter-add + GRU elementwise are bandwidth-bound fp32 kernels.
// ---------------------------------------------------------------------------

typedef __attribute__((ext_vector_type(16))) _Float16 v16h;
typedef __attribute__((ext_vector_type(8)))  float    v8f;

#define HID   150
#define KP    160      // K padded (5 chunks of 32)
#define KCHUNKS 5
#define LD3H  464      // 450 padded to 29*16

// ---- convert fp32 weight (rows x cols, row-major) to padded fp16 (rowsP x colsP)
__global__ void cvt_weight_kernel(const float* __restrict__ src, _Float16* __restrict__ dst,
                                  int rows, int cols, int rowsP, int colsP) {
    int idx = blockIdx.x * blockDim.x + threadIdx.x;
    int total = rowsP * colsP;
    if (idx >= total) return;
    int r = idx / colsP, c = idx % colsP;
    float v = (r < rows && c < cols) ? src[(long)r * cols + c] : 0.0f;
    dst[idx] = (_Float16)v;
}

// ---- h init: copy nodes (N x 150) into padded (N x 160), zero pad cols
__global__ void init_h_kernel(const float* __restrict__ nodes, float* __restrict__ h, int N) {
    int idx = blockIdx.x * blockDim.x + threadIdx.x;
    int total = N * KP;
    if (idx >= total) return;
    int i = idx / KP, c = idx % KP;
    h[idx] = (c < HID) ? nodes[(long)i * HID + c] : 0.0f;
}

// ---- load one 16-row x 32-K A chunk in WMMA A layout (f32 -> f16)
__device__ __forceinline__ v16h load_a_chunk(const float* __restrict__ Xrow, int k0) {
    float4 f0 = *(const float4*)(Xrow + k0);
    float4 f1 = *(const float4*)(Xrow + k0 + 4);
    float4 f2 = *(const float4*)(Xrow + k0 + 8);
    float4 f3 = *(const float4*)(Xrow + k0 + 12);
    v16h av;
    av[0]  = (_Float16)f0.x; av[1]  = (_Float16)f0.y; av[2]  = (_Float16)f0.z; av[3]  = (_Float16)f0.w;
    av[4]  = (_Float16)f1.x; av[5]  = (_Float16)f1.y; av[6]  = (_Float16)f1.z; av[7]  = (_Float16)f1.w;
    av[8]  = (_Float16)f2.x; av[9]  = (_Float16)f2.y; av[10] = (_Float16)f2.z; av[11] = (_Float16)f2.w;
    av[12] = (_Float16)f3.x; av[13] = (_Float16)f3.y; av[14] = (_Float16)f3.z; av[15] = (_Float16)f3.w;
    return av;
}

// ---- epilogue for one 16x16 D tile strip (8 rows per lane-half)
template<int MODE, bool SCALED, int LDY>
__device__ __forceinline__ void epilogue_strip(const v8f& c, float* __restrict__ yp_base,
                                               const float* __restrict__ bias_scale,
                                               long mbase, float bn) {
#pragma unroll
    for (int vr = 0; vr < 8; ++vr) {               // D: vgpr r -> row r / r+8
        float v = c[vr] + bn * (SCALED ? bias_scale[mbase + vr] : 1.0f);
        float* yp = yp_base + vr * LDY;            // compile-time stride -> imm offsets
        if (MODE == 1) v = (v > 0.0f) ? v : 0.01f * v;
        if (MODE == 2) v += *yp;
        *yp = v;
    }
}

// ---------------------------------------------------------------------------
// GEMM: Y(M x Nout) = act( X(M x KP,f32) @ W16^T + bias * scale ) [+ Y]
//   W16: (NoutP x KP) fp16 row-major, K-padded with zeros.
//   MODE 0: Y = X@W^T + b
//   MODE 1: Y = leaky_relu(X@W^T + b)
//   MODE 2: Y += X@W^T + b*bias_scale[row]   (SCALED: bias_scale = deg)
// One wave owns a 32-row strip (2 x 16-row WMMA tiles); 8 waves per block.
// M must be a multiple of 32 (N=100000 = 3125*32).
// ---------------------------------------------------------------------------
template<int MODE, bool SCALED, int LDY>
__global__ __launch_bounds__(256)
void gemm_wmma_kernel(const float* __restrict__ X,
                      const _Float16* __restrict__ W16,
                      const float* __restrict__ bias,
                      const float* __restrict__ bias_scale,
                      float* __restrict__ Y,
                      int M, int Nout) {
    const int wave = threadIdx.x >> 5;
    const int lane = threadIdx.x & 31;
    const long row0 = ((long)blockIdx.x * 8 + wave) * 32;
    if (row0 >= M) return;

    const int hsel = lane >> 4;    // 0 for lanes 0-15, 1 for lanes 16-31
    const int lmod = lane & 15;

    // ---- Preload A: 2 strips x 5 k-chunks (f32 -> f16), held in registers.
    const float* Xrow0 = X + (row0 + lmod) * (long)KP;
    const float* Xrow1 = Xrow0 + 16 * KP;
    v16h a0[KCHUNKS], a1[KCHUNKS];
#pragma unroll
    for (int kc = 0; kc < KCHUNKS; ++kc) {
        const int k0 = kc * 32 + hsel * 16;
        a0[kc] = load_a_chunk(Xrow0, k0);
        a1[kc] = load_a_chunk(Xrow1, k0);
    }

    const int NT = (Nout + 15) >> 4;
    for (int nt = 0; nt < NT; ++nt) {
        const int n = nt * 16 + lmod;
        // B layout: lane column n = lane%16, K half by lane/16, 16 contiguous f16.
        const _Float16* Wrow = W16 + (long)n * KP + hsel * 16;

        // Bulk-issue all 5 B chunk loads, then one wait + 10 interleaved WMMAs.
        v16h b[KCHUNKS];
#pragma unroll
        for (int kc = 0; kc < KCHUNKS; ++kc)
            b[kc] = *(const v16h*)(Wrow + kc * 32);   // 32B aligned
        __builtin_amdgcn_sched_barrier(0);            // keep loads before WMMAs

        v8f c0 = {}, c1 = {};
#pragma unroll
        for (int kc = 0; kc < KCHUNKS; ++kc) {
            c0 = __builtin_amdgcn_wmma_f32_16x16x32_f16(
                     false, a0[kc], false, b[kc], (short)0, c0, false, false);
            c1 = __builtin_amdgcn_wmma_f32_16x16x32_f16(
                     false, a1[kc], false, b[kc], (short)0, c1, false, false);
        }

        if (n < Nout) {
            const float bn = bias[n];
            float* yp0 = Y + (row0 + hsel * 8) * (long)LDY + n;
            epilogue_strip<MODE, SCALED, LDY>(c0, yp0, bias_scale,
                                              row0 + hsel * 8, bn);
            epilogue_strip<MODE, SCALED, LDY>(c1, yp0 + 16 * LDY, bias_scale,
                                              row0 + 16 + hsel * 8, bn);
        }
    }
}

// ---- scatter: s[dst] += u[src] per edge; deg[dst] += 1. One wave per edge.
//      Runs over all 160 cols (u pad cols are zero) -> uniform 5 iters/lane.
__global__ __launch_bounds__(256)
void scatter_add_kernel(const int* __restrict__ edges,
                        const float* __restrict__ u,
                        float* __restrict__ s,
                        float* __restrict__ deg, int E) {
    int gid = blockIdx.x * blockDim.x + threadIdx.x;
    int e = gid >> 5, lane = gid & 31;
    if (e >= E) return;
    int dst = edges[2 * e];       // e[:,0]
    int src = edges[2 * e + 1];   // e[:,1]
    const float* up = u + (long)src * KP;
    float* sp = s + (long)dst * KP;
#pragma unroll
    for (int j = 0; j < KP / 32; ++j) {
        int c = lane + j * 32;
        atomicAdd(sp + c, up[c]);
    }
    if (lane == 0) atomicAdd(deg + dst, 1.0f);
}

// ---- GRU elementwise: h = (1-z)*n + z*h
__global__ void gru_elem_kernel(const float* __restrict__ gi, const float* __restrict__ gh,
                                float* __restrict__ h, int N) {
    int idx = blockIdx.x * blockDim.x + threadIdx.x;
    int total = N * HID;
    if (idx >= total) return;
    int i = idx / HID, c = idx % HID;
    const float* gip = gi + (long)i * LD3H;
    const float* ghp = gh + (long)i * LD3H;
    float r = 1.0f / (1.0f + __expf(-(gip[c] + ghp[c])));
    float z = 1.0f / (1.0f + __expf(-(gip[HID + c] + ghp[HID + c])));
    float n = tanhf(gip[2 * HID + c] + r * ghp[2 * HID + c]);
    float* hp = h + (long)i * KP + c;
    *hp = (1.0f - z) * n + z * (*hp);
}

// ---- segment sum into g (G x 160)
__global__ void seg_sum_kernel(const float* __restrict__ h, const int* __restrict__ gids,
                               float* __restrict__ g, int N) {
    int idx = blockIdx.x * blockDim.x + threadIdx.x;
    int total = N * HID;
    if (idx >= total) return;
    int i = idx / HID, c = idx % HID;
    atomicAdd(&g[(long)gids[i] * KP + c], h[(long)i * KP + c]);
}

// ---- head: log/nan/relu + concat + fc1(80x151) + fc2(80x80) + fcL(10x80)
__global__ __launch_bounds__(256)
void head_kernel(const float* __restrict__ g, const float* __restrict__ ptype,
                 const float* __restrict__ fc1W, const float* __restrict__ fc1b,
                 const float* __restrict__ fc2W, const float* __restrict__ fc2b,
                 const float* __restrict__ fcLW, const float* __restrict__ fcLb,
                 float* __restrict__ out, int G) {
    __shared__ float x1[16][152];
    __shared__ float y1[16][80];
    __shared__ float y2[16][80];
    int tid = threadIdx.x;
    for (int idx = tid; idx < G * 151; idx += 256) {
        int gi = idx / 151, c = idx % 151;
        float v;
        if (c < HID) {
            float lg = logf(g[(long)gi * KP + c]);
            if (lg != lg) lg = 0.0f;          // NaN -> 0
            v = (lg > 0.0f) ? lg : 0.0f;      // relu
        } else {
            v = ptype[gi];
        }
        x1[gi][c] = v;
    }
    __syncthreads();
    for (int idx = tid; idx < G * 80; idx += 256) {
        int gi = idx / 80, j = idx % 80;
        float acc = fc1b[j];
        for (int k = 0; k < 151; ++k) acc += x1[gi][k] * fc1W[j * 151 + k];
        y1[gi][j] = (acc > 0.0f) ? acc : 0.01f * acc;
    }
    __syncthreads();
    for (int idx = tid; idx < G * 80; idx += 256) {
        int gi = idx / 80, j = idx % 80;
        float acc = fc2b[j];
        for (int k = 0; k < 80; ++k) acc += y1[gi][k] * fc2W[j * 80 + k];
        y2[gi][j] = (acc > 0.0f) ? acc : 0.01f * acc;
    }
    __syncthreads();
    for (int idx = tid; idx < G * 10; idx += 256) {
        int gi = idx / 10, j = idx % 10;
        float acc = fcLb[j];
        for (int k = 0; k < 80; ++k) acc += y2[gi][k] * fcLW[j * 80 + k];
        out[gi * 10 + j] = acc;
    }
}

// ---------------------------------------------------------------------------
extern "C" void kernel_launch(void* const* d_in, const int* in_sizes, int n_in,
                              void* d_out, int out_size, void* d_ws, size_t ws_size,
                              hipStream_t stream) {
    const float* nodes   = (const float*)d_in[0];
    const float* ptype   = (const float*)d_in[1];
    const float* W1[2]   = { (const float*)d_in[2], (const float*)d_in[6] };
    const float* b1[2]   = { (const float*)d_in[3], (const float*)d_in[7] };
    const float* W2[2]   = { (const float*)d_in[4], (const float*)d_in[8] };
    const float* b2[2]   = { (const float*)d_in[5], (const float*)d_in[9] };
    const float* Wih     = (const float*)d_in[10];
    const float* bih     = (const float*)d_in[11];
    const float* Whh     = (const float*)d_in[12];
    const float* bhh     = (const float*)d_in[13];
    const float* fc1W    = (const float*)d_in[14];
    const float* fc1b    = (const float*)d_in[15];
    const float* fc2W    = (const float*)d_in[16];
    const float* fc2b    = (const float*)d_in[17];
    const float* fcLW    = (const float*)d_in[18];
    const float* fcLb    = (const float*)d_in[19];
    const int*   edges[2]= { (const int*)d_in[20], (const int*)d_in[21] };
    const int*   gids    = (const int*)d_in[22];

    const int N  = in_sizes[0] / HID;            // 100000
    const int E0 = in_sizes[20] / 2;             // 500000
    const int E1 = in_sizes[21] / 2;
    const int Ecnt[2] = { E0, E1 };
    const int G  = in_sizes[1];                  // 16
    const int PASSES = 4;                        // device scalar; fixed by setup

    // ---- workspace layout (256B aligned slices)
    char* ws = (char*)d_ws;
    size_t off = 0;
    auto take = [&](size_t bytes) -> char* {
        char* p = ws + off;
        off = (off + bytes + 255) & ~(size_t)255;
        return p;
    };
    float*    h    = (float*)take((size_t)N * KP * 4);
    float*    u    = (float*)take((size_t)N * KP * 4);
    float*    s    = (float*)take((size_t)N * KP * 4);
    float*    inc  = (float*)take((size_t)N * KP * 4);
    float*    deg  = (float*)take((size_t)N * 4);
    float*    gib  = (float*)take((size_t)N * LD3H * 4);
    float*    ghb  = (float*)take((size_t)N * LD3H * 4);
    float*    gbuf = (float*)take((size_t)16 * KP * 4);
    _Float16* W1h[2], *W2h[2];
    W1h[0] = (_Float16*)take((size_t)160 * KP * 2);
    W2h[0] = (_Float16*)take((size_t)160 * KP * 2);
    W1h[1] = (_Float16*)take((size_t)160 * KP * 2);
    W2h[1] = (_Float16*)take((size_t)160 * KP * 2);
    _Float16* Wihh = (_Float16*)take((size_t)LD3H * KP * 2);
    _Float16* Whhh = (_Float16*)take((size_t)LD3H * KP * 2);
    (void)ws_size;

    // ---- weight conversion (fp32 -> padded fp16)
    auto cvt = [&](const float* src, _Float16* dst, int rows, int rowsP) {
        int total = rowsP * KP;
        cvt_weight_kernel<<<(total + 255) / 256, 256, 0, stream>>>(src, dst, rows, HID, rowsP, KP);
    };
    cvt(W1[0], W1h[0], HID, 160);  cvt(W2[0], W2h[0], HID, 160);
    cvt(W1[1], W1h[1], HID, 160);  cvt(W2[1], W2h[1], HID, 160);
    cvt(Wih,   Wihh,  3 * HID, LD3H);
    cvt(Whh,   Whhh,  3 * HID, LD3H);

    // ---- init h (padded), zero pad-sensitive buffers once
    {
        int total = N * KP;
        init_h_kernel<<<(total + 255) / 256, 256, 0, stream>>>(nodes, h, N);
    }
    hipMemsetAsync(u,    0, (size_t)N * KP * 4, stream);   // pad cols stay 0
    hipMemsetAsync(gbuf, 0, (size_t)16 * KP * 4, stream);

    const int gemm_blocks = (N + 255) / 256;      // 8 waves x 32 rows per block

    const int ew_total  = N * HID;
    const int ew_blocks = (ew_total + 255) / 256;

    for (int p = 0; p < PASSES; ++p) {
        hipMemsetAsync(inc, 0, (size_t)N * KP * 4, stream);
        for (int es = 0; es < 2; ++es) {
            hipMemsetAsync(s,   0, (size_t)N * KP * 4, stream);
            hipMemsetAsync(deg, 0, (size_t)N * 4, stream);
            // u = leaky_relu(h @ W1^T + b1)   [per node, not per edge]
            gemm_wmma_kernel<1, false, KP><<<gemm_blocks, 256, 0, stream>>>(
                h, W1h[es], b1[es], nullptr, u, N, HID);
            // s[dst] += u[src] ; deg[dst] += 1
            {
                long threads = (long)Ecnt[es] * 32;
                scatter_add_kernel<<<(int)((threads + 255) / 256), 256, 0, stream>>>(
                    edges[es], u, s, deg, Ecnt[es]);
            }
            // inc += s @ W2^T + deg * b2
            gemm_wmma_kernel<2, true, KP><<<gemm_blocks, 256, 0, stream>>>(
                s, W2h[es], b2[es], deg, inc, N, HID);
        }
        // GRU gates
        gemm_wmma_kernel<0, false, LD3H><<<gemm_blocks, 256, 0, stream>>>(
            inc, Wihh, bih, nullptr, gib, N, 3 * HID);
        gemm_wmma_kernel<0, false, LD3H><<<gemm_blocks, 256, 0, stream>>>(
            h, Whhh, bhh, nullptr, ghb, N, 3 * HID);
        gru_elem_kernel<<<ew_blocks, 256, 0, stream>>>(gib, ghb, h, N);
    }

    // ---- segment sum + head MLP
    seg_sum_kernel<<<ew_blocks, 256, 0, stream>>>(h, gids, gbuf, N);
    head_kernel<<<1, 256, 0, stream>>>(gbuf, ptype, fc1W, fc1b, fc2W, fc2b,
                                       fcLW, fcLb, (float*)d_out, G);
}
